// GemNet_78408922956493
// MI455X (gfx1250) — matrix operand order repaired and checked by reading, not compiled, not run on hardware
//
#include <hip/hip_runtime.h>

typedef _Float16 half_t;
typedef __attribute__((ext_vector_type(16))) _Float16 v16h;
typedef __attribute__((ext_vector_type(8)))  _Float16 v8h;
typedef __attribute__((ext_vector_type(8)))  float    v8f;

static constexpr int cNA = 10000;
static constexpr int cNE = 100000;
static constexpr int cNT = 300000;
static constexpr float INV_SQRT2F = 0.70710678118654752f;
static constexpr float CUTOFFF = 5.0f;

// ---------------------------------------------------------------------------
// f16 WMMA GEMM:  C[M,N] = act( A[M,K] @ Bt[N,K]^T )
// Block tile 64(M) x 64(N), 256 threads = 8 waves (4 row x 2 col).
// Each wave owns a 16(M) x 32(N) strip -> 2 x v_wmma_f32_16x16x32_f16 per
// K-step. Double-buffered LDS, one barrier per K-step, branch-free loads
// (row clamp; garbage rows only feed stores that the M-guard discards).
// Fragment layouts per CDNA5 ISA 7.12.2 (16-bit A 16x32, B 32x16, f32 C/D).
// ---------------------------------------------------------------------------
template <bool ACT_SILU>
__global__ __launch_bounds__(256) void k_gemm(const half_t* __restrict__ A,
                                              const half_t* __restrict__ Bt,
                                              half_t* __restrict__ C,
                                              int M, int N, int K) {
  __shared__ __attribute__((aligned(32))) half_t As[2][64][40];
  __shared__ __attribute__((aligned(32))) half_t Bs[2][64][40];
  const int tid  = threadIdx.x;
  const int lane = tid & 31;
  const int wave = tid >> 5;
  const int wrow = wave >> 1;      // 0..3 : M strip
  const int wcol = wave & 1;       // 0..1 : N strip (x32)
  const int hsel = lane >> 4;      // 0/1
  const int l15  = lane & 15;
  const int row0 = blockIdx.x * 64;
  const int col0 = blockIdx.y * 64;

  // staging: thread loads one v8h of A and one v8h of B per K-step
  const int lr = tid >> 2;          // 0..63
  const int lc = (tid & 3) * 8;     // 0,8,16,24
  int arow = row0 + lr; if (arow > M - 1) arow = M - 1;   // clamp: branch-free
  const half_t* pA = A + (size_t)arow * K + lc;
  const half_t* pB = Bt + (size_t)(col0 + lr) * K + lc;

  v8f acc0 = {}, acc1 = {};

  auto compute = [&](int cb) {
    v8h alo = *(const v8h*)&As[cb][wrow * 16 + l15][hsel * 8];
    v8h ahi = *(const v8h*)&As[cb][wrow * 16 + l15][16 + hsel * 8];
    v8h b0l = *(const v8h*)&Bs[cb][wcol * 32 + l15][hsel * 8];
    v8h b0h = *(const v8h*)&Bs[cb][wcol * 32 + l15][16 + hsel * 8];
    v8h b1l = *(const v8h*)&Bs[cb][wcol * 32 + 16 + l15][hsel * 8];
    v8h b1h = *(const v8h*)&Bs[cb][wcol * 32 + 16 + l15][16 + hsel * 8];
    v16h af, bf0, bf1;
#pragma unroll
    for (int i = 0; i < 8; ++i) {
      af[i] = alo[i];  af[8 + i] = ahi[i];
      bf0[i] = b0l[i]; bf0[8 + i] = b0h[i];
      bf1[i] = b1l[i]; bf1[8 + i] = b1h[i];
    }
    acc0 = __builtin_amdgcn_wmma_f32_16x16x32_f16(false, af, false, bf0,
                                                  (short)0, acc0, false, false);
    acc1 = __builtin_amdgcn_wmma_f32_16x16x32_f16(false, af, false, bf1,
                                                  (short)0, acc1, false, false);
  };

  // preload tile 0
  {
    v8h ga = *(const v8h*)pA;
    v8h gb = *(const v8h*)pB;
    *(v8h*)&As[0][lr][lc] = ga;
    *(v8h*)&Bs[0][lr][lc] = gb;
  }
  __syncthreads();

  int cur = 0;
  for (int k0 = 32; k0 < K; k0 += 32) {
    // issue next-tile global loads first (latency overlapped with WMMA below)
    v8h na = *(const v8h*)(pA + k0);
    v8h nb = *(const v8h*)(pB + k0);
    if (k0 + 32 < K) __builtin_prefetch(pA + k0 + 32, 0, 0);  // global_prefetch_b8
    compute(cur);
    *(v8h*)&As[cur ^ 1][lr][lc] = na;
    *(v8h*)&Bs[cur ^ 1][lr][lc] = nb;
    __syncthreads();
    cur ^= 1;
  }
  compute(cur);

  const int ccol = col0 + wcol * 32 + l15;
#pragma unroll
  for (int v = 0; v < 8; ++v) {
    const int crow = row0 + wrow * 16 + hsel * 8 + v;  // C/D: VGPR v -> M = v + hsel*8
    if (crow < M) {
      float x0 = acc0[v], x1 = acc1[v];
      if (ACT_SILU) {
        x0 = x0 / (1.0f + __expf(-x0));
        x1 = x1 / (1.0f + __expf(-x1));
      }
      C[(size_t)crow * N + ccol]      = (half_t)x0;
      C[(size_t)crow * N + ccol + 16] = (half_t)x1;
    }
  }
}

// ---------------------------------------------------------------------------
// Elementwise / scatter / prep kernels
// ---------------------------------------------------------------------------
__global__ void k_zero_f32(float* p, size_t n) {
  size_t i = (size_t)blockIdx.x * 256 + threadIdx.x;
  if (i < n) p[i] = 0.f;
}

__global__ void k_wt(const float* __restrict__ W, half_t* __restrict__ Wt,
                     int K, int N, int Kp) {
  size_t i = (size_t)blockIdx.x * 256 + threadIdx.x;
  if (i >= (size_t)N * Kp) return;
  int n = (int)(i / Kp), k = (int)(i % Kp);
  Wt[i] = (k < K) ? (half_t)W[(size_t)k * N + n] : (half_t)0.f;
}

__global__ void k_bilT(const float* __restrict__ bil, half_t* __restrict__ bt) {
  int i = blockIdx.x * 256 + threadIdx.x;          // 64 * 1024
  if (i >= 64 * 1024) return;
  int o = i >> 10, k = i & 1023, c = k >> 6, j = k & 63;
  bt[i] = (half_t)bil[(size_t)j * 1024 + c * 64 + o];   // bil[j][c][o]
}

__global__ void k_geom(const float* __restrict__ R, const int* __restrict__ is,
                       const int* __restrict__ it, float* __restrict__ V,
                       float* __restrict__ rbf) {
  int e = blockIdx.x * 256 + threadIdx.x;
  if (e >= cNE) return;
  int s = is[e], t = it[e];
  float vx = R[t * 3] - R[s * 3];
  float vy = R[t * 3 + 1] - R[s * 3 + 1];
  float vz = R[t * 3 + 2] - R[s * 3 + 2];
  float D = sqrtf(vx * vx + vy * vy + vz * vz);
  float inv = 1.0f / fmaxf(D, 1e-6f);
  V[e * 3] = vx * inv; V[e * 3 + 1] = vy * inv; V[e * 3 + 2] = vz * inv;
  float d = fmaxf(D, 1e-6f) / CUTOFFF;
  float d2 = d * d, d4 = d2 * d2, d5 = d4 * d, d6 = d5 * d, d7 = d6 * d;
  float env = (d < 1.0f) ? (1.0f / d - 28.f * d5 + 48.f * d6 - 21.f * d7) : 0.f;
  const float norm = 0.6324555320336759f;  // sqrt(2/cutoff)
  for (int n = 0; n < 6; ++n)
    rbf[e * 6 + n] = env * norm * sinf(3.14159265358979f * (n + 1) * d);
}

__global__ __launch_bounds__(256) void k_rbfsmall(
    const float* __restrict__ rbf, const float* __restrict__ w3,
    const float* __restrict__ wh, const float* __restrict__ wo,
    const float* __restrict__ cbf, half_t* __restrict__ r3,
    half_t* __restrict__ rh, half_t* __restrict__ ro, float* __restrict__ rW1) {
  __shared__ float s3[96], sh[96], so[96], sc[672];
  for (int i = threadIdx.x; i < 96; i += 256) { s3[i] = w3[i]; sh[i] = wh[i]; so[i] = wo[i]; }
  for (int i = threadIdx.x; i < 672; i += 256) sc[i] = cbf[i];
  __syncthreads();
  int e = blockIdx.x * 256 + threadIdx.x;
  if (e >= cNE) return;
  float r[6];
  for (int n = 0; n < 6; ++n) r[n] = rbf[(size_t)e * 6 + n];
  for (int c = 0; c < 16; ++c) {
    float a = 0, b = 0, g = 0;
    for (int n = 0; n < 6; ++n) {
      a += r[n] * s3[n * 16 + c]; b += r[n] * sh[n * 16 + c]; g += r[n] * so[n * 16 + c];
    }
    r3[(size_t)e * 32 + c] = (half_t)a;
    rh[(size_t)e * 32 + c] = (half_t)b;
    ro[(size_t)e * 32 + c] = (half_t)g;
  }
  for (int c = 16; c < 32; ++c) {   // zero pad K 16 -> 32
    r3[(size_t)e * 32 + c] = (half_t)0.f;
    rh[(size_t)e * 32 + c] = (half_t)0.f;
    ro[(size_t)e * 32 + c] = (half_t)0.f;
  }
  for (int l = 0; l < 7; ++l)
    for (int c = 0; c < 16; ++c) {
      float a = 0;
      for (int n = 0; n < 6; ++n) a += r[n] * sc[l * 96 + n * 16 + c];
      rW1[(size_t)e * 112 + l * 16 + c] = a;
    }
}

__global__ void k_sph(const float* __restrict__ V, const int* __restrict__ red,
                      const int* __restrict__ expd, float* __restrict__ sph) {
  int t = blockIdx.x * 256 + threadIdx.x;
  if (t >= cNT) return;
  int e1 = red[t], e2 = expd[t];
  float ax = V[e1 * 3], ay = V[e1 * 3 + 1], az = V[e1 * 3 + 2];
  float bx = V[e2 * 3], by = V[e2 * 3 + 1], bz = V[e2 * 3 + 2];
  float cosx = ax * bx + ay * by + az * bz;
  float cx = ay * bz - az * by, cy = az * bx - ax * bz, cz = ax * by - ay * bx;
  float siny = fmaxf(sqrtf(cx * cx + cy * cy + cz * cz), 1e-9f);
  float x = cosf(atan2f(siny, cosx));
  float pm2 = 1.f, pm1 = x;
  sph[(size_t)t * 7] = pm2; sph[(size_t)t * 7 + 1] = pm1;
  for (int l = 2; l < 7; ++l) {
    float p = ((2 * l - 1) * x * pm1 - (l - 1) * pm2) / (float)l;
    sph[(size_t)t * 7 + l] = p; pm2 = pm1; pm1 = p;
  }
}

__global__ void k_embed(const float* __restrict__ emb, const int* __restrict__ Z,
                        half_t* __restrict__ h) {
  int i = blockIdx.x * 256 + threadIdx.x;
  if (i >= cNA * 128) return;
  int a = i >> 7, c = i & 127;
  h[i] = (half_t)emb[(size_t)Z[a] * 128 + c];
}

__global__ void k_concat_emb(const half_t* __restrict__ h, const float* __restrict__ rbf,
                             const int* __restrict__ is, const int* __restrict__ it,
                             half_t* __restrict__ out) {
  size_t i = (size_t)blockIdx.x * 256 + threadIdx.x;
  if (i >= (size_t)cNE * 288) return;
  int e = (int)(i / 288), c = (int)(i % 288);
  half_t v;
  if (c < 128)      v = h[(size_t)is[e] * 128 + c];
  else if (c < 256) v = h[(size_t)it[e] * 128 + (c - 128)];
  else if (c < 262) v = (half_t)rbf[(size_t)e * 6 + (c - 256)];
  else              v = (half_t)0.f;
  out[i] = v;
}

__global__ void k_concat_cat(const half_t* __restrict__ h, const half_t* __restrict__ m,
                             const int* __restrict__ is, const int* __restrict__ it,
                             half_t* __restrict__ out) {
  size_t i = (size_t)blockIdx.x * 256 + threadIdx.x;
  if (i >= (size_t)cNE * 512) return;
  int e = (int)(i >> 9), c = (int)(i & 511);
  half_t v;
  if (c < 128)      v = h[(size_t)is[e] * 128 + c];
  else if (c < 256) v = h[(size_t)it[e] * 128 + (c - 128)];
  else              v = m[(size_t)e * 256 + (c - 256)];
  out[i] = v;
}

// x[i] *= y[i] ; vectorized 8 halves per thread (n % 8 == 0)
__global__ void k_mul8(half_t* __restrict__ x, const half_t* __restrict__ y, size_t n8) {
  size_t i = (size_t)blockIdx.x * 256 + threadIdx.x;
  if (i >= n8) return;
  v8h a = *(v8h*)(x + i * 8);
  v8h b = *(const v8h*)(y + i * 8);
#pragma unroll
  for (int k = 0; k < 8; ++k) a[k] = (half_t)((float)a[k] * (float)b[k]);
  *(v8h*)(x + i * 8) = a;
}

// x[i] = (x[i] + y[i]) * s ; vectorized (n % 8 == 0)
__global__ void k_addscale8(half_t* __restrict__ x, const half_t* __restrict__ y,
                            float s, size_t n8) {
  size_t i = (size_t)blockIdx.x * 256 + threadIdx.x;
  if (i >= n8) return;
  v8h a = *(v8h*)(x + i * 8);
  v8h b = *(const v8h*)(y + i * 8);
#pragma unroll
  for (int k = 0; k < 8; ++k) a[k] = (half_t)(((float)a[k] + (float)b[k]) * s);
  *(v8h*)(x + i * 8) = a;
}

// a[e][:] = (a[e][:] + b[swap[e]][:]) * s ; one thread per 8 cols
__global__ void k_swap_add(half_t* __restrict__ a, const half_t* __restrict__ b,
                           const int* __restrict__ swp) {
  size_t i = (size_t)blockIdx.x * 256 + threadIdx.x;   // over NE*32 groups
  if (i >= (size_t)cNE * 32) return;
  int e = (int)(i >> 5), g = (int)(i & 31);
  v8h av = *(v8h*)(a + ((size_t)e << 8) + g * 8);
  v8h bv = *(const v8h*)(b + ((size_t)swp[e] << 8) + g * 8);
#pragma unroll
  for (int k = 0; k < 8; ++k)
    av[k] = (half_t)(((float)av[k] + (float)bv[k]) * INV_SQRT2F);
  *(v8h*)(a + ((size_t)e << 8) + g * 8) = av;
}

__global__ void k_mul_scatter(const half_t* __restrict__ m, const half_t* __restrict__ g,
                              const int* __restrict__ it, float* __restrict__ X) {
  size_t i = (size_t)blockIdx.x * 256 + threadIdx.x;
  if (i >= (size_t)cNE * 256) return;
  int e = (int)(i >> 8), c = (int)(i & 255);
  atomicAdd(&X[(size_t)it[e] * 256 + c], (float)m[i] * (float)g[i]);
}

__global__ void k_f2h(const float* __restrict__ x, half_t* __restrict__ y, size_t n) {
  size_t i = (size_t)blockIdx.x * 256 + threadIdx.x;
  if (i < n) y[i] = (half_t)x[i];
}

__global__ void k_trip_scatter(const half_t* __restrict__ xba2, const float* __restrict__ sph,
                               const int* __restrict__ expd, const int* __restrict__ red,
                               float* __restrict__ sumk) {
  size_t i = (size_t)blockIdx.x * 256 + threadIdx.x;
  if (i >= (size_t)cNT * 64) return;
  int t = (int)(i >> 6), j = (int)(i & 63);
  float x = (float)xba2[(size_t)expd[t] * 64 + j];
  size_t base = (size_t)red[t] * 448 + j;
#pragma unroll
  for (int l = 0; l < 7; ++l)
    atomicAdd(&sumk[base + (size_t)l * 64], sph[(size_t)t * 7 + l] * x);
}

__global__ __launch_bounds__(256) void k_tmp(const float* __restrict__ rW1,
                                             const float* __restrict__ sumk,
                                             half_t* __restrict__ tmpH, int e0) {
  __shared__ float W[112];
  __shared__ float S[448];
  int e = e0 + blockIdx.x;
  for (int i = threadIdx.x; i < 112; i += 256) W[i] = rW1[(size_t)e * 112 + i];
  for (int i = threadIdx.x; i < 448; i += 256) S[i] = sumk[(size_t)e * 448 + i];
  __syncthreads();
  for (int i = threadIdx.x; i < 1024; i += 256) {
    int c = i >> 6, j = i & 63;
    float a = 0;
#pragma unroll
    for (int l = 0; l < 7; ++l) a += W[l * 16 + c] * S[l * 64 + j];
    tmpH[(size_t)blockIdx.x * 1024 + i] = (half_t)a;   // tmp[e][c][j], chunk-local
  }
}

__global__ __launch_bounds__(256) void k_final(const half_t* __restrict__ x,
                                               const float* __restrict__ Wf,
                                               float* __restrict__ out) {
  __shared__ float red[256];
  int a = blockIdx.x * 256 + threadIdx.x;
  float s = 0.f;
  if (a < cNA)
    for (int c = 0; c < 128; ++c) s += (float)x[(size_t)a * 128 + c] * Wf[c];
  red[threadIdx.x] = s;
  __syncthreads();
  for (int st = 128; st > 0; st >>= 1) {
    if (threadIdx.x < st) red[threadIdx.x] += red[threadIdx.x + st];
    __syncthreads();
  }
  if (threadIdx.x == 0) atomicAdd(out, red[0]);
}

// ---------------------------------------------------------------------------
static inline unsigned nb(size_t n) { return (unsigned)((n + 255) / 256); }

extern "C" void kernel_launch(void* const* d_in, const int* in_sizes, int n_in,
                              void* d_out, int out_size, void* d_ws, size_t ws_size,
                              hipStream_t stream) {
  (void)in_sizes; (void)n_in; (void)ws_size;
  // Leaf order: jax tree_leaves (sorted dict keys at every level):
  //   R, Z, id3_expand_ba, id3_reduce_ca, id_swap, idx_s, idx_t, params-leaves
  const float* R       = (const float*)d_in[0];
  const int*  Z        = (const int*)d_in[1];
  const int*  id3_exp  = (const int*)d_in[2];
  const int*  id3_red  = (const int*)d_in[3];
  const int*  id_swap  = (const int*)d_in[4];
  const int*  idx_s    = (const int*)d_in[5];
  const int*  idx_t    = (const int*)d_in[6];
  const float* const* P = (const float* const*)&d_in[7];
  // params (sorted): 0 atom_emb, 1 edge_emb, 2..21 int0, 22..41 int1, 42..61 int2,
  //   62 mlp_cbf3, 63 mlp_rbf3, 64 mlp_rbf_h, 65 mlp_rbf_out, 66.. out0..out3 (7 each)
  // int block leaf order (sorted): as0_w1,as0_w2,au0_w1,au0_w2,au1_w1,au1_w2,
  //   au_first,au_rbf,bil,bs0_w1,bs0_w2,cc0_w1,cc0_w2,concat_w,dense_ba,dense_ca,
  //   down,trip_rbf,up_ac,up_ca
  // out block leaf order (sorted): final,first,r0_w1,r0_w2,r1_w1,r1_w2,rbf

  char* wsb = (char*)d_ws;
  size_t off = 0;
  auto alloc  = [&](size_t bytes) -> void* {
    off = (off + 255) & ~(size_t)255; void* p = wsb + off; off += bytes; return p;
  };
  auto allocH = [&](size_t e) -> half_t* { return (half_t*)alloc(e * sizeof(half_t)); };
  auto allocF = [&](size_t e) -> float* { return (float*)alloc(e * sizeof(float)); };

  // activations
  float*  Vv    = allocF((size_t)cNE * 3);
  float*  rbfF  = allocF((size_t)cNE * 6);
  half_t* rbf3h = allocH((size_t)cNE * 32);
  half_t* rbfhh = allocH((size_t)cNE * 32);
  half_t* rbfoh = allocH((size_t)cNE * 32);
  float*  rW1   = allocF((size_t)cNE * 112);
  float*  sphF  = allocF((size_t)cNT * 7);
  half_t* hA    = allocH((size_t)cNA * 128);
  half_t* xa0   = allocH((size_t)cNA * 128);
  half_t* xa1   = allocH((size_t)cNA * 128);
  half_t* xa2   = allocH((size_t)cNA * 128);
  float*  XatomF = allocF((size_t)cNA * 256);
  half_t* XatomH = allocH((size_t)cNA * 256);
  half_t* bufM  = allocH((size_t)cNE * 256);
  half_t* bufA  = allocH((size_t)cNE * 256);
  half_t* bufB  = allocH((size_t)cNE * 256);
  half_t* bufC  = allocH((size_t)cNE * 256);
  half_t* bufD  = allocH((size_t)cNE * 256);
  half_t* bufCat = allocH((size_t)cNE * 512);
  half_t* xba2  = allocH((size_t)cNE * 64);
  half_t* xbil  = allocH((size_t)cNE * 64);
  float*  sumk  = allocF((size_t)cNE * 448);
  const int CHUNK = 25000;
  half_t* tmpH  = allocH((size_t)CHUNK * 1024);

  auto wt = [&](const float* W, int K, int N, int Kp) -> half_t* {
    half_t* p = allocH((size_t)N * Kp);
    k_wt<<<nb((size_t)N * Kp), 256, 0, stream>>>(W, p, K, N, Kp);
    return p;
  };
  auto gemm = [&](const half_t* A, const half_t* Bt, half_t* C, int M, int N, int K,
                  bool silu) {
    dim3 g((unsigned)((M + 63) / 64), (unsigned)(N / 64));
    if (silu) k_gemm<true><<<g, 256, 0, stream>>>(A, Bt, C, M, N, K);
    else      k_gemm<false><<<g, 256, 0, stream>>>(A, Bt, C, M, N, K);
  };

  // ---- weight prep (f32 K x N  ->  f16 N x Kpad) -------------------------
  half_t* WtEdge = wt(P[1], 262, 256, 288);
  struct OutW { const float* fin; half_t *first, *r[4], *rbf; } ow[4];
  for (int b = 0; b < 4; ++b) {
    const float* const* q = &P[66 + 7 * b];
    ow[b].fin   = q[0];
    ow[b].first = wt(q[1], 256, 128, 256);
    for (int i = 0; i < 4; ++i) ow[b].r[i] = wt(q[2 + i], 128, 128, 128);
    ow[b].rbf   = wt(q[6], 16, 256, 32);
  }
  struct IntW { half_t *as_[2], *au[4], *auFirst, *auRbf, *bilT, *bs_[2], *cc[2],
                *concat, *ba, *ca, *down, *trip, *upAc, *upCa; } iw[3];
  for (int b = 0; b < 3; ++b) {
    const float* const* q = &P[2 + 20 * b];
    iw[b].as_[0] = wt(q[0], 256, 256, 256);
    iw[b].as_[1] = wt(q[1], 256, 256, 256);
    for (int i = 0; i < 4; ++i) iw[b].au[i] = wt(q[2 + i], 128, 128, 128);
    iw[b].auFirst = wt(q[6], 256, 128, 256);
    iw[b].auRbf   = wt(q[7], 16, 256, 32);
    iw[b].bilT    = allocH((size_t)64 * 1024);
    k_bilT<<<nb((size_t)64 * 1024), 256, 0, stream>>>(q[8], iw[b].bilT);
    iw[b].bs_[0] = wt(q[9], 256, 256, 256);
    iw[b].bs_[1] = wt(q[10], 256, 256, 256);
    iw[b].cc[0]  = wt(q[11], 256, 256, 256);
    iw[b].cc[1]  = wt(q[12], 256, 256, 256);
    iw[b].concat = wt(q[13], 512, 256, 512);
    iw[b].ba     = wt(q[14], 256, 256, 256);
    iw[b].ca     = wt(q[15], 256, 256, 256);
    iw[b].down   = wt(q[16], 256, 64, 256);
    iw[b].trip   = wt(q[17], 16, 256, 32);
    iw[b].upAc   = wt(q[18], 64, 256, 64);
    iw[b].upCa   = wt(q[19], 64, 256, 64);
  }

  // ---- geometry / bases / embedding --------------------------------------
  k_zero_f32<<<1, 32, 0, stream>>>((float*)d_out, (size_t)out_size);
  k_geom<<<nb(cNE), 256, 0, stream>>>(R, idx_s, idx_t, Vv, rbfF);
  k_rbfsmall<<<nb(cNE), 256, 0, stream>>>(rbfF, P[63], P[64], P[65], P[62],
                                          rbf3h, rbfhh, rbfoh, rW1);
  k_sph<<<nb(cNT), 256, 0, stream>>>(Vv, id3_red, id3_exp, sphF);
  k_embed<<<nb((size_t)cNA * 128), 256, 0, stream>>>(P[0], Z, hA);
  k_concat_emb<<<nb((size_t)cNE * 288), 256, 0, stream>>>(hA, rbfF, idx_s, idx_t, bufCat);
  gemm(bufCat, WtEdge, bufM, cNE, 256, 288, true);   // m = silu(concat @ edge_emb)
  half_t* m = bufM;

  auto out_block = [&](const OutW& o, half_t* mcur) {
    gemm(rbfoh, o.rbf, bufD, cNE, 256, 32, false);   // Gout = rbf_out @ p.rbf
    k_zero_f32<<<nb((size_t)cNA * 256), 256, 0, stream>>>(XatomF, (size_t)cNA * 256);
    k_mul_scatter<<<nb((size_t)cNE * 256), 256, 0, stream>>>(mcur, bufD, idx_t, XatomF);
    k_f2h<<<nb((size_t)cNA * 256), 256, 0, stream>>>(XatomF, XatomH, (size_t)cNA * 256);
    gemm(XatomH, o.first, xa0, cNA, 128, 256, true);
    for (int i = 0; i < 2; ++i) {
      gemm(xa0, o.r[2 * i],     xa1, cNA, 128, 128, true);
      gemm(xa1, o.r[2 * i + 1], xa2, cNA, 128, 128, true);
      k_addscale8<<<nb((size_t)cNA * 16), 256, 0, stream>>>(xa0, xa2, INV_SQRT2F,
                                                            (size_t)cNA * 16);
    }
    k_final<<<nb(cNA), 256, 0, stream>>>(xa0, o.fin, (float*)d_out);
  };

  out_block(ow[0], m);

  for (int b = 0; b < 3; ++b) {
    IntW& w = iw[b];
    // --- triplet interaction ---
    gemm(m, w.ca, bufA, cNE, 256, 256, true);              // x_skip
    gemm(m, w.ba, bufB, cNE, 256, 256, true);
    gemm(rbf3h, w.trip, bufC, cNE, 256, 32, false);
    k_mul8<<<nb((size_t)cNE * 32), 256, 0, stream>>>(bufB, bufC, (size_t)cNE * 32);
    gemm(bufB, w.down, xba2, cNE, 64, 256, true);          // x_ba
    k_zero_f32<<<nb((size_t)cNE * 448), 256, 0, stream>>>(sumk, (size_t)cNE * 448);
    k_trip_scatter<<<nb((size_t)cNT * 64), 256, 0, stream>>>(xba2, sphF, id3_exp,
                                                             id3_red, sumk);
    for (int c0 = 0; c0 < cNE; c0 += CHUNK) {
      int mc = (cNE - c0 < CHUNK) ? (cNE - c0) : CHUNK;
      k_tmp<<<mc, 256, 0, stream>>>(rW1, sumk, tmpH, c0);
      gemm(tmpH, w.bilT, xbil + (size_t)c0 * 64, mc, 64, 1024, false);  // x_bil
    }
    gemm(xbil, w.upCa, bufB, cNE, 256, 64, true);
    gemm(xbil, w.upAc, bufC, cNE, 256, 64, true);
    k_swap_add<<<nb((size_t)cNE * 32), 256, 0, stream>>>(bufB, bufC, id_swap);  // x3
    k_addscale8<<<nb((size_t)cNE * 32), 256, 0, stream>>>(bufA, bufB, INV_SQRT2F,
                                                          (size_t)cNE * 32);   // x=(skip+x3)*s
    gemm(bufA, w.bs_[0], bufB, cNE, 256, 256, true);       // residual (before skip)
    gemm(bufB, w.bs_[1], bufC, cNE, 256, 256, true);
    k_addscale8<<<nb((size_t)cNE * 32), 256, 0, stream>>>(bufA, bufC, INV_SQRT2F,
                                                          (size_t)cNE * 32);
    k_addscale8<<<nb((size_t)cNE * 32), 256, 0, stream>>>(bufA, m, INV_SQRT2F,
                                                          (size_t)cNE * 32);   // +m skip
    gemm(bufA, w.as_[0], bufB, cNE, 256, 256, true);       // residual (after skip)
    gemm(bufB, w.as_[1], bufC, cNE, 256, 256, true);
    k_addscale8<<<nb((size_t)cNE * 32), 256, 0, stream>>>(bufA, bufC, INV_SQRT2F,
                                                          (size_t)cNE * 32);
    { half_t* t = m; m = bufA; bufA = t; }                 // m = x

    // --- atom update ---
    gemm(rbfhh, w.auRbf, bufB, cNE, 256, 32, false);
    k_zero_f32<<<nb((size_t)cNA * 256), 256, 0, stream>>>(XatomF, (size_t)cNA * 256);
    k_mul_scatter<<<nb((size_t)cNE * 256), 256, 0, stream>>>(m, bufB, idx_t, XatomF);
    k_f2h<<<nb((size_t)cNA * 256), 256, 0, stream>>>(XatomF, XatomH, (size_t)cNA * 256);
    gemm(XatomH, w.auFirst, xa0, cNA, 128, 256, true);
    for (int i = 0; i < 2; ++i) {
      gemm(xa0, w.au[2 * i],     xa1, cNA, 128, 128, true);
      gemm(xa1, w.au[2 * i + 1], xa2, cNA, 128, 128, true);
      k_addscale8<<<nb((size_t)cNA * 16), 256, 0, stream>>>(xa0, xa2, INV_SQRT2F,
                                                            (size_t)cNA * 16);
    }
    k_addscale8<<<nb((size_t)cNA * 16), 256, 0, stream>>>(hA, xa0, INV_SQRT2F,
                                                          (size_t)cNA * 16);   // h update

    // --- concat / edge refresh ---
    k_concat_cat<<<nb((size_t)cNE * 512), 256, 0, stream>>>(hA, m, idx_s, idx_t, bufCat);
    gemm(bufCat, w.concat, bufB, cNE, 256, 512, true);
    gemm(bufB, w.cc[0], bufC, cNE, 256, 256, true);
    gemm(bufC, w.cc[1], bufD, cNE, 256, 256, true);
    k_addscale8<<<nb((size_t)cNE * 32), 256, 0, stream>>>(bufB, bufD, INV_SQRT2F,
                                                          (size_t)cNE * 32);
    { half_t* t = m; m = bufB; bufB = t; }                 // m = refreshed edges

    out_block(ow[b + 1], m);
  }
}
